// ConcentrationNeuron_80041010528789
// MI455X (gfx1250) — matrix operand order repaired and checked by reading, not compiled, not run on hardware
//
#include <hip/hip_runtime.h>

// ---------------------------------------------------------------------------
// ConcentrationNeuron on MI455X (gfx1250, wave32).
//
//   1) embed_delta : pt = x@e_w+e_b (fp32 WMMA 16x16x4), delta = wrap2pi(pt)
//   2) 3-pass chunked prefix-sum of delta over S with spinor wrap -> hist (ph)
//   3) readout     : logits = feats@r_w+r_b via f16 WMMA 16x16x32 (fp32 acc).
//                    Block = 128(M)x128(N); B K-chunks staged to LDS with
//                    GLOBAL_LOAD_ASYNC_TO_LDS_B128 (ASYNCcnt), double-buffered;
//                    each lane's 64 needed ph values live in VGPRs.
//   4) parity      : tiny fp32 GEMV
// ---------------------------------------------------------------------------

typedef __attribute__((ext_vector_type(16))) _Float16 v16h;
typedef __attribute__((ext_vector_type(8)))  float    v8f;
typedef __attribute__((ext_vector_type(2)))  float    v2f;

#define NB   32
#define NS   4096
#define ND   64
#define NH   128
#define NOUT 512
#define NM   (NB*NS)      // 131072 rows, m = b*NS + s
#define NCHK 64           // scan chunks per sequence
#define CLEN 64           // steps per chunk (NCHK*CLEN == NS)

#define PI_F          3.14159265358979323846f
#define TWO_PI_F      6.28318530717958647692f
#define FOUR_PI_F     12.5663706143591729539f
#define INV_TWO_PI_F  0.15915494309189533577f
#define INV_FOUR_PI_F 0.07957747154594766788f

// --- gfx1250 async global->LDS copy helpers (ASYNCcnt tracked) -------------
// INST_OFFSET applies to both the LDS and global address (ISA 15.18.3), so a
// single (lds_off, gsrc) pair with offsets 0/16 moves 32 contiguous bytes.
__device__ __forceinline__ void async_copy32(unsigned lds_off,
                                             const unsigned int* gsrc) {
    asm volatile("global_load_async_to_lds_b128 %0, %1, off\n\t"
                 "global_load_async_to_lds_b128 %0, %1, off offset:16"
                 :: "v"(lds_off), "v"(gsrc)
                 : "memory");
}
__device__ __forceinline__ void wait_async0() {
    asm volatile("s_wait_asynccnt 0" ::: "memory");
}
// Truncate a generic pointer to LDS to its 32-bit LDS byte address
// (flat LDS address = {shared aperture, offset[31:0]}).
__device__ __forceinline__ unsigned lds_off_of(const void* p) {
    return (unsigned)(unsigned long long)p;
}

// ---------------------------------------------------------------------------
// 1) Embed GEMM + delta.  Wave computes a 16(M) x 128(N) tile, K=64 in steps
//    of 4 via V_WMMA_F32_16X16X4_F32 (fp32 kept: delta feeds a long scan).
// ---------------------------------------------------------------------------
__global__ __launch_bounds__(128)
void embed_delta_kernel(const float* __restrict__ x,    // (NM, 64)
                        const float* __restrict__ e_w,  // (64, 128)
                        const float* __restrict__ e_b,  // (128)
                        float* __restrict__ delta)      // (NM, 128)
{
    const int lane = threadIdx.x & 31;
    const int wave = threadIdx.x >> 5;
    const int mtile = blockIdx.x * 4 + wave;
    const int row0 = mtile * 16;
    const bool hi = lane >= 16;
    const int n0 = lane & 15;
    const int arow = row0 + n0;

    v8f acc[8];
#pragma unroll
    for (int i = 0; i < 8; ++i) acc[i] = (v8f)0.0f;

#pragma unroll 4
    for (int k = 0; k < ND; k += 4) {
        // A fragment (16x4 f32): VGPR0 = K+0 (lo) / K+2 (hi); VGPR1 = +1
        const float* ap = x + (size_t)arow * ND + k + (hi ? 2 : 0);
        v2f a; a.x = ap[0]; a.y = ap[1];
        const int kk = k + (hi ? 2 : 0);
#pragma unroll
        for (int nt = 0; nt < 8; ++nt) {
            const int col = nt * 16 + n0;
            // B fragment (4x16 f32): VGPR j -> row K=kk+j, N=lane%16
            v2f b;
            b.x = e_w[(size_t)kk * NH + col];
            b.y = e_w[(size_t)(kk + 1) * NH + col];
            acc[nt] = __builtin_amdgcn_wmma_f32_16x16x4_f32(
                false, a, false, b, (short)0, acc[nt], false, false);
        }
    }

    // Epilogue: add bias, wrap to (-pi, pi]  (== atan2(sin pt, cos pt))
#pragma unroll
    for (int nt = 0; nt < 8; ++nt) {
        const int col = nt * 16 + n0;
        const float bias = e_b[col];
#pragma unroll
        for (int v = 0; v < 8; ++v) {
            const int row = row0 + v + (hi ? 8 : 0);
            float pt = acc[nt][v] + bias;
            float d = pt - TWO_PI_F * rintf(pt * INV_TWO_PI_F);
            delta[(size_t)row * NH + col] = d;
        }
    }
}

// ---------------------------------------------------------------------------
// 2a) Per-chunk partial sums of delta (no wrap; |sum| <= 64*pi, fine in f32)
// ---------------------------------------------------------------------------
__global__ __launch_bounds__(256)
void scan_chunks_kernel(const float* __restrict__ delta,  // (NM, NH)
                        float* __restrict__ csum)         // (NB*NH, NCHK)
{
    const int t = blockIdx.x * 256 + threadIdx.x;   // < NB*NCHK*NH
    const int h = t & (NH - 1);
    const int bc = t >> 7;
    const int c = bc & (NCHK - 1);
    const int b = bc >> 6;
    const float* dp = delta + ((size_t)(b * NS + c * CLEN)) * NH + h;
    float s = 0.0f;
#pragma unroll 8
    for (int i = 0; i < CLEN; ++i) s += dp[(size_t)i * NH];
    csum[((size_t)(b * NH + h)) * NCHK + c] = s;
}

// ---------------------------------------------------------------------------
// 2b) Exclusive prefix over chunk sums, wrapped to [-2pi, 2pi) each step
// ---------------------------------------------------------------------------
__global__ __launch_bounds__(256)
void scan_offsets_kernel(const float* __restrict__ csum,  // (NB*NH, NCHK)
                         const float* __restrict__ ph_in, // (NB, NH)
                         float* __restrict__ offs)        // (NB*NH, NCHK)
{
    const int t = blockIdx.x * 256 + threadIdx.x;   // < NB*NH
    float acc = ph_in[t];
    acc -= FOUR_PI_F * floorf((acc + TWO_PI_F) * INV_FOUR_PI_F);
    const float* cp = csum + (size_t)t * NCHK;
    float* op = offs + (size_t)t * NCHK;
    for (int c = 0; c < NCHK; ++c) {
        op[c] = acc;
        acc += cp[c];
        acc -= FOUR_PI_F * floorf((acc + TWO_PI_F) * INV_FOUR_PI_F);
    }
}

// ---------------------------------------------------------------------------
// 2c) Apply: serial 64-step wrapped accumulation inside each chunk -> hist
// ---------------------------------------------------------------------------
__global__ __launch_bounds__(256)
void scan_apply_kernel(const float* __restrict__ delta, // (NM, NH)
                       const float* __restrict__ offs,  // (NB*NH, NCHK)
                       float* __restrict__ hist)        // (NM, NH) == ph
{
    const int t = blockIdx.x * 256 + threadIdx.x;   // < NB*NCHK*NH
    const int h = t & (NH - 1);
    const int bc = t >> 7;
    const int c = bc & (NCHK - 1);
    const int b = bc >> 6;
    float acc = offs[((size_t)(b * NH + h)) * NCHK + c];
    const float* dp = delta + ((size_t)(b * NS + c * CLEN)) * NH + h;
    float* hp = hist + ((size_t)(b * NS + c * CLEN)) * NH + h;
#pragma unroll 4
    for (int i = 0; i < CLEN; ++i) {
        acc += dp[(size_t)i * NH];               // |acc+d| < 3pi
        if (acc >= TWO_PI_F)       acc -= FOUR_PI_F;
        else if (acc < -TWO_PI_F)  acc += FOUR_PI_F;
        hp[(size_t)i * NH] = acc;
    }
}

// ---------------------------------------------------------------------------
// 3a) Pack r_w (640x512 fp32) into f16 WMMA B fragments.
//     Fragment (kc, nt): lane l holds N = nt*16 + l%16, K halves
//     kc*32 + (l>=16 ? 8 : 0) + 2j + (j>=4 ? 8 : 0) + {0,1} in dword j (j=0..7).
// ---------------------------------------------------------------------------
__global__ __launch_bounds__(256)
void pack_rw_kernel(const float* __restrict__ r_w,   // (640, 512)
                    unsigned int* __restrict__ rwp)  // 20*32 frags * 32 lanes * 8 dw
{
    const int t = blockIdx.x * 256 + threadIdx.x;    // < 163840
    const int j = t & 7;
    const int lane = (t >> 3) & 31;
    const int frag = t >> 8;
    const int kc = frag >> 5;
    const int nt = frag & 31;
    const int k = kc * 32 + ((lane >= 16) ? 8 : 0) + 2 * j + ((j >= 4) ? 8 : 0);
    const int n = nt * 16 + (lane & 15);
    const _Float16 h0 = (_Float16)r_w[(size_t)k * NOUT + n];
    const _Float16 h1 = (_Float16)r_w[(size_t)(k + 1) * NOUT + n];
    union { _Float16 h[2]; unsigned int u; } pk;
    pk.h[0] = h0; pk.h[1] = h1;
    rwp[t] = pk.u;
}

// ---------------------------------------------------------------------------
// Feature transform: f in {cos, sin, cos/2, sin/2, identity}
// ---------------------------------------------------------------------------
__device__ __forceinline__ float feat_xform(int f, float ph) {
    if      (f == 0) return __cosf(ph);
    else if (f == 1) return __sinf(ph);
    else if (f == 2) return __cosf(0.5f * ph);
    else if (f == 3) return __sinf(0.5f * ph);
    return ph;
}

// ---------------------------------------------------------------------------
// 3b) Readout GEMM: 256 threads = 8 waves, block tile 128(M) x 128(N).
//     Wave w owns rows [blk*128 + w*16, +16), all 8 N-tiles of the block.
//     K=640 in 20 chunks of 32 via V_WMMA_F32_16X16X32_F16.  Per-chunk B
//     (8 frags = 8KB) is async-staged into LDS, double-buffered, shared by
//     all 8 waves.  A (feats) generated from 64 register-resident ph values:
//     feature f = kc/4, h = (kc%4)*32 + klocal.
//     A f16 16x32 lane layout: lo lanes K={0..7,16..23}, hi K={8..15,24..31}.
// ---------------------------------------------------------------------------
__global__ __launch_bounds__(256)
void readout_kernel(const float* __restrict__ hist,       // (NM, 128) ph
                    const unsigned int* __restrict__ rwp, // packed B frags
                    const float* __restrict__ r_b,        // (512)
                    float* __restrict__ logits)           // (NM, 512)
{
    __shared__ __attribute__((aligned(32))) unsigned int bbuf[2][2048]; // 2x8KB

    const int t = threadIdx.x;
    const int lane = t & 31;
    const int wave = t >> 5;                 // 0..7 = M sub-tile
    const int row0 = blockIdx.x * 128 + wave * 16;
    const int by = blockIdx.y;               // N group: cols [by*128, +128)
    const bool hi = lane >= 16;
    const int n0 = lane & 15;
    const int arow = row0 + n0;

    // ---- preload this lane's 64 ph values into VGPRs ----
    // phr[q*8+e] = ph[arow][q*16 + (hi?8:0) + e]
    float phr[64];
    {
        const float* base = hist + (size_t)arow * NH + (hi ? 8 : 0);
#pragma unroll
        for (int q = 0; q < 8; ++q)
#pragma unroll
            for (int e = 0; e < 8; ++e)
                phr[q * 8 + e] = base[q * 16 + e];
    }

    // ---- async-stage B chunk 0 (8KB: thread t copies 32B at t*32) ----
    async_copy32(lds_off_of(&bbuf[0][t * 8]),
                 rwp + (size_t)(by * 8) * 256 + t * 8);

    v8f acc[8];
#pragma unroll
    for (int i = 0; i < 8; ++i) acc[i] = (v8f)0.0f;

#pragma unroll
    for (int kc = 0; kc < 20; ++kc) {
        const int cur = kc & 1;
        // wait for this chunk's async stores to LDS, publish across waves
        wait_async0();
        __syncthreads();
        // prefetch next chunk into the other buffer
        if (kc + 1 < 20) {
            async_copy32(lds_off_of(&bbuf[cur ^ 1][t * 8]),
                         rwp + (size_t)(((kc + 1) * 32 + by * 8)) * 256 + t * 8);
        }

        // Build A fragment from register-resident ph
        const int f = kc >> 2;
        const int q0 = (kc & 3) * 2;
        v16h a;
#pragma unroll
        for (int e = 0; e < 8; ++e) {
            a[e]     = (_Float16)feat_xform(f, phr[q0 * 8 + e]);       // K 0..7
            a[8 + e] = (_Float16)feat_xform(f, phr[(q0 + 1) * 8 + e]); // K 16..23
        }

#pragma unroll
        for (int ntl = 0; ntl < 8; ++ntl) {
            const v16h bfrag =
                *reinterpret_cast<const v16h*>(&bbuf[cur][(ntl * 32 + lane) * 8]);
            acc[ntl] = __builtin_amdgcn_wmma_f32_16x16x32_f16(
                false, a, false, bfrag, (short)0, acc[ntl], false, false);
        }
    }

    // Epilogue: + r_b, scatter C (VGPR v -> row v or v+8; N = lane%16)
#pragma unroll
    for (int ntl = 0; ntl < 8; ++ntl) {
        const int col = by * 128 + ntl * 16 + n0;
        const float bias = r_b[col];
#pragma unroll
        for (int v = 0; v < 8; ++v) {
            const int row = row0 + v + (hi ? 8 : 0);
            logits[(size_t)row * NOUT + col] = acc[ntl][v] + bias;
        }
    }
}

// ---------------------------------------------------------------------------
// 4) Parity GEMV (K=640), fp32 VALU; half-angle identities reuse one sincos.
// ---------------------------------------------------------------------------
__global__ __launch_bounds__(256)
void parity_kernel(const float* __restrict__ hist,  // (NM, 128)
                   const float* __restrict__ p_w,   // (640)
                   const float* __restrict__ p_b,   // (1)
                   float* __restrict__ parity)      // (NM)
{
    const int m = blockIdx.x * 256 + threadIdx.x;
    if (m >= NM) return;
    const float* pp = hist + (size_t)m * NH;
    float s = p_b[0];
#pragma unroll 4
    for (int h = 0; h < NH; ++h) {
        const float ph = pp[h];
        const float ch = __cosf(0.5f * ph);
        const float sh = __sinf(0.5f * ph);
        const float c  = 2.0f * ch * ch - 1.0f;   // cos(ph)
        const float sn = 2.0f * sh * ch;          // sin(ph)
        s += c  * p_w[h]
           + sn * p_w[NH + h]
           + ch * p_w[2 * NH + h]
           + sh * p_w[3 * NH + h]
           + ph * p_w[4 * NH + h];
    }
    parity[m] = s;
}

// ---------------------------------------------------------------------------
extern "C" void kernel_launch(void* const* d_in, const int* in_sizes, int n_in,
                              void* d_out, int out_size, void* d_ws, size_t ws_size,
                              hipStream_t stream) {
    const float* x     = (const float*)d_in[0];
    const float* ph_in = (const float*)d_in[1];
    const float* e_w   = (const float*)d_in[2];
    const float* e_b   = (const float*)d_in[3];
    const float* r_w   = (const float*)d_in[4];
    const float* r_b   = (const float*)d_in[5];
    const float* p_w   = (const float*)d_in[6];
    const float* p_b   = (const float*)d_in[7];

    float* out    = (float*)d_out;
    float* logits = out;                               // (B,S,OUT) = (NM,512)
    float* parity = logits + (size_t)NM * NOUT;        // (NM,)
    float* hist   = parity + (size_t)NM;               // (B,S,H) = (NM,128)

    float* delta = (float*)d_ws;                       // NM*NH f32   (64 MB)
    float* csum  = delta + (size_t)NM * NH;            // NB*NH*NCHK  (1 MB)
    float* offs  = csum + (size_t)NB * NH * NCHK;      // NB*NH*NCHK  (1 MB)
    unsigned int* rwp = (unsigned int*)(offs + (size_t)NB * NH * NCHK); // 640 KB

    pack_rw_kernel    <<<640,               256, 0, stream>>>(r_w, rwp);
    embed_delta_kernel<<<NM / 64,           128, 0, stream>>>(x, e_w, e_b, delta);
    scan_chunks_kernel<<<(NB*NH*NCHK)/256,  256, 0, stream>>>(delta, csum);
    scan_offsets_kernel<<<(NB*NH)/256,      256, 0, stream>>>(csum, ph_in, offs);
    scan_apply_kernel <<<(NB*NH*NCHK)/256,  256, 0, stream>>>(delta, offs, hist);
    readout_kernel    <<<dim3(NM/128, NOUT/128), 256, 0, stream>>>(hist, rwp, r_b, logits);
    parity_kernel     <<<NM / 256,          256, 0, stream>>>(hist, p_w, p_b, parity);
}
